// NGCF_66718021976721
// MI455X (gfx1250) — matrix operand order repaired and checked by reading, not compiled, not run on hardware
//
#include <hip/hip_runtime.h>
#include <hip/hip_bf16.h>

typedef __attribute__((ext_vector_type(16))) _Float16 v16h;
typedef __attribute__((ext_vector_type(8)))  _Float16 v8h;
typedef __attribute__((ext_vector_type(4)))  _Float16 v4h;
typedef __attribute__((ext_vector_type(8)))  float    v8f;

#define USER_NUM  100000
#define ITEM_NUM  200000
#define N_NODES   (USER_NUM + ITEM_NUM)
#define EMB_DIM   64
#define NNZ       2000000
#define BATCH     4096
#define NEG_SLOPE 0.2f
#define DECAY     0.0001f
#define INV_KEEP  (1.0f / 0.9f)
// floor(0.9 * 2^32)
#define KEEP_THRESH 3865470566u

__device__ __forceinline__ unsigned wang_hash(unsigned x) {
    x = (x ^ 61u) ^ (x >> 16);
    x *= 9u;
    x ^= x >> 4;
    x *= 0x27d4eb2du;
    x ^= x >> 15;
    return x;
}

// ---- init: edge dropout + zero accumulators ------------------------------
__global__ void k_edge_drop(const float* __restrict__ edge_vals,
                            float* __restrict__ vals,
                            float* __restrict__ accum) {
    int gid = blockIdx.x * blockDim.x + threadIdx.x;
    if (gid == 0) { accum[0] = 0.f; accum[1] = 0.f; accum[2] = 0.f; accum[3] = 0.f; }
    if (gid >= NNZ) return;
    unsigned h = wang_hash((unsigned)gid * 2654435761u ^ 0xC0FFEE42u);
    vals[gid] = (h < KEEP_THRESH) ? edge_vals[gid] * INV_KEEP : 0.0f;
}

// ---- build ego = concat(user_emb, item_emb) ------------------------------
__global__ void k_build_ego(const float* __restrict__ user_emb,
                            const float* __restrict__ item_emb,
                            float* __restrict__ ego) {
    int gid = blockIdx.x * blockDim.x + threadIdx.x;     // < 19.2M, fits i32
    if (gid >= N_NODES * EMB_DIM) return;
    const int usz = USER_NUM * EMB_DIM;
    ego[gid] = (gid < usz) ? user_emb[gid] : item_emb[gid - usz];
}

// ---- zero a float buffer --------------------------------------------------
__global__ void k_zero(float* __restrict__ p, int n) {
    int gid = blockIdx.x * blockDim.x + threadIdx.x;
    if (gid < n) p[gid] = 0.0f;
}

// ---- scatter: side[row] += vals[e] * ego[col]  (float2 per thread) -------
__global__ void k_scatter(const int* __restrict__ edge_row,
                          const int* __restrict__ edge_col,
                          const float* __restrict__ vals,
                          const float* __restrict__ ego,
                          float* __restrict__ side) {
    unsigned gid = blockIdx.x * blockDim.x + threadIdx.x;   // < 64M, fits u32
    if (gid >= (unsigned)NNZ * 32u) return;
    int e = (int)(gid >> 5);
    int d = (int)(gid & 31u) * 2;
    float v = vals[e];
    if (v == 0.0f) return;
    int col = edge_col[e];
    int row = edge_row[e];
    const float2 x = *(const float2*)(ego + col * EMB_DIM + d);
    float* dst = side + row * EMB_DIM + d;
    unsafeAtomicAdd(dst + 0, x.x * v);
    unsafeAtomicAdd(dst + 1, x.y * v);
}

// ---- fused dual-GEMM (WMMA) + bias + leaky-relu + msg dropout ------------
// ego_out = drop(leaky_relu(side @ Wg + bg + (ego*side) @ Wb + bb))
// 256 threads (8 waves). Block covers 128 rows; wave w owns 16-row tile w,
// all 64 output cols (4 col-tiles x 2 K-steps x 2 GEMMs = 16 WMMAs).
// rowBase is clamped so the last block overlaps its predecessor instead of
// needing tail guards: overlapped rows recompute identical values.
#define ASTR 72   // padded LDS stride (halfs) for A/P tiles   [row][k]
#define WSTR 72   // padded LDS stride (halfs) for W^T tiles   [n][k]
#define RPB  128  // rows per block

union Frag16 { v16h v; v8h h[2]; };

__global__ __launch_bounds__(256)
void k_gemm_act(const float* __restrict__ ego,
                const float* __restrict__ side,
                const float* __restrict__ Wg,
                const float* __restrict__ bg,
                const float* __restrict__ Wb,
                const float* __restrict__ bb,
                float* __restrict__ ego_out,
                unsigned seed) {
    __shared__ alignas(16) _Float16 sWgT[64 * WSTR];   // [n][k]
    __shared__ alignas(16) _Float16 sWbT[64 * WSTR];   // [n][k]
    __shared__ alignas(16) _Float16 sA[RPB * ASTR];    // f16(side)      [row][k]
    __shared__ alignas(16) _Float16 sP[RPB * ASTR];    // f16(ego*side)  [row][k]

    const int tid = threadIdx.x;

    // stage weights transposed: sW*T[n*WSTR + k] = W[k*64 + n]
    for (int i = tid; i < 4096; i += 256) {
        int k = i >> 6, n = i & 63;             // coalesced global read over n
        sWgT[n * WSTR + k] = (_Float16)Wg[i];
        sWbT[n * WSTR + k] = (_Float16)Wb[i];
    }

    // clamped row base: no tail guards anywhere (overlap is deterministic)
    int rowBase = blockIdx.x * RPB;
    if (rowBase > N_NODES - RPB) rowBase = N_NODES - RPB;

    // stage A tiles: 128 rows x 64 cols, float4 loads -> 4-half packed stores
    for (int i = tid; i < RPB * 16; i += 256) {
        int r = i >> 4;
        int c = (i & 15) * 4;
        int idx = (rowBase + r) * EMB_DIM + c;
        const float4 s = *(const float4*)(side + idx);
        const float4 e = *(const float4*)(ego  + idx);
        v4h hs, hp;
        hs[0] = (_Float16)s.x; hs[1] = (_Float16)s.y;
        hs[2] = (_Float16)s.z; hs[3] = (_Float16)s.w;
        hp[0] = (_Float16)(s.x * e.x); hp[1] = (_Float16)(s.y * e.y);
        hp[2] = (_Float16)(s.z * e.z); hp[3] = (_Float16)(s.w * e.w);
        *(v4h*)(sA + r * ASTR + c) = hs;
        *(v4h*)(sP + r * ASTR + c) = hp;
    }
    __syncthreads();

    const int wave  = tid >> 5;
    const int lane  = tid & 31;
    const int mBase = wave * 16;                 // block-relative row tile
    const int m     = mBase + (lane & 15);       // A row this lane supplies
    const int abase = (lane < 16) ? 0 : 8;       // A K-subblock (16x32 layout)
    const int bbase = (lane < 16) ? 0 : 8;       // B K-subblock (32x16 layout)
    const int dRow0 = rowBase + mBase + ((lane < 16) ? 0 : 8);  // D row for r=0

    // A/P fragments: fixed per wave, reused across all 4 col-tiles
    Frag16 a0, a1, p0, p1;
    a0.h[0] = *(const v8h*)(sA + m * ASTR +  0 + abase);
    a0.h[1] = *(const v8h*)(sA + m * ASTR + 16 + abase);
    a1.h[0] = *(const v8h*)(sA + m * ASTR + 32 + abase);
    a1.h[1] = *(const v8h*)(sA + m * ASTR + 48 + abase);
    p0.h[0] = *(const v8h*)(sP + m * ASTR +  0 + abase);
    p0.h[1] = *(const v8h*)(sP + m * ASTR + 16 + abase);
    p1.h[0] = *(const v8h*)(sP + m * ASTR + 32 + abase);
    p1.h[1] = *(const v8h*)(sP + m * ASTR + 48 + abase);

    #pragma unroll
    for (int ct = 0; ct < 4; ++ct) {
        const int n = ct * 16 + (lane & 15);     // output column this lane owns
        Frag16 wg0, wg1, wb0, wb1;               // two 8-half K runs each (same order as A)
        wg0.h[0] = *(const v8h*)(sWgT + n * WSTR +  0 + bbase);
        wg0.h[1] = *(const v8h*)(sWgT + n * WSTR + 16 + bbase);
        wg1.h[0] = *(const v8h*)(sWgT + n * WSTR + 32 + bbase);
        wg1.h[1] = *(const v8h*)(sWgT + n * WSTR + 48 + bbase);
        wb0.h[0] = *(const v8h*)(sWbT + n * WSTR +  0 + bbase);
        wb0.h[1] = *(const v8h*)(sWbT + n * WSTR + 16 + bbase);
        wb1.h[0] = *(const v8h*)(sWbT + n * WSTR + 32 + bbase);
        wb1.h[1] = *(const v8h*)(sWbT + n * WSTR + 48 + bbase);

        v8f acc_g = {};
        v8f acc_b = {};
        acc_g = __builtin_amdgcn_wmma_f32_16x16x32_f16(
            false, a0.v, false, wg0.v, (short)0, acc_g, false, false);
        acc_g = __builtin_amdgcn_wmma_f32_16x16x32_f16(
            false, a1.v, false, wg1.v, (short)0, acc_g, false, false);
        acc_b = __builtin_amdgcn_wmma_f32_16x16x32_f16(
            false, p0.v, false, wb0.v, (short)0, acc_b, false, false);
        acc_b = __builtin_amdgcn_wmma_f32_16x16x32_f16(
            false, p1.v, false, wb1.v, (short)0, acc_b, false, false);

        const float bsum = bg[n] + bb[n];
        const unsigned hbase = ((unsigned)dRow0 * 64u + (unsigned)n);
        float* outp = ego_out + dRow0 * EMB_DIM + n;
        #pragma unroll
        for (int r = 0; r < 8; ++r) {
            float t = acc_g[r] + acc_b[r] + bsum;
            t = (t > 0.f) ? t : NEG_SLOPE * t;             // leaky relu
            unsigned h = wang_hash((hbase + (unsigned)r * 64u) * 2246822519u ^ seed);
            t = (h < KEEP_THRESH) ? t * INV_KEEP : 0.f;
            outp[r * EMB_DIM] = t;
        }
    }
}

// ---- gather batch rows (+optional row L2-normalize) into feature block ---
__global__ void k_gather_norm(const float* __restrict__ src,
                              const int* __restrict__ ids,
                              int id_offset,
                              float* __restrict__ dst,  // BATCH x 256
                              int col_block,            // 0..3
                              int do_norm) {
    int b = blockIdx.x * blockDim.x + threadIdx.x;
    if (b >= BATCH) return;
    int node = ids[b] + id_offset;
    const float* r = src + node * EMB_DIM;
    float scale = 1.0f;
    if (do_norm) {
        float ss = 0.f;
        #pragma unroll
        for (int i = 0; i < EMB_DIM; ++i) { float x = r[i]; ss += x * x; }
        scale = 1.0f / fmaxf(sqrtf(ss), 1e-12f);
    }
    float* o = dst + b * 256 + col_block * EMB_DIM;
    #pragma unroll
    for (int i = 0; i < EMB_DIM; ++i) o[i] = r[i] * scale;
}

// ---- BPR loss + reg, block reduce, atomic accumulate ----------------------
__global__ void k_loss(const float* __restrict__ u,
                       const float* __restrict__ p,
                       const float* __restrict__ nf,
                       float* __restrict__ accum) {
    __shared__ float smf[256];
    __shared__ float srg[256];
    int b = blockIdx.x * blockDim.x + threadIdx.x;
    float mf = 0.f, reg = 0.f;
    if (b < BATCH) {
        const float* uu = u  + b * 256;
        const float* pp = p  + b * 256;
        const float* nn = nf + b * 256;
        float ps = 0.f, ns = 0.f;
        #pragma unroll 8
        for (int i = 0; i < 256; ++i) {
            float uv = uu[i], pv = pp[i], nv = nn[i];
            ps  += uv * pv;
            ns  += uv * nv;
            reg += uv * uv + pv * pv + nv * nv;
        }
        float x = ps - ns;
        // -log_sigmoid(x) = softplus(-x), numerically stable
        mf = (x > 0.f) ? log1pf(expf(-x)) : (-x + log1pf(expf(x)));
    }
    smf[threadIdx.x] = mf;
    srg[threadIdx.x] = reg;
    __syncthreads();
    for (int s = 128; s > 0; s >>= 1) {
        if (threadIdx.x < s) {
            smf[threadIdx.x] += smf[threadIdx.x + s];
            srg[threadIdx.x] += srg[threadIdx.x + s];
        }
        __syncthreads();
    }
    if (threadIdx.x == 0) {
        unsafeAtomicAdd(&accum[0], smf[0]);
        unsafeAtomicAdd(&accum[1], srg[0]);
    }
}

__global__ void k_finalize(const float* __restrict__ accum,
                           float* __restrict__ out) {
    if (threadIdx.x == 0 && blockIdx.x == 0) {
        float mf_loss  = accum[0] / (float)BATCH;
        float emb_loss = DECAY * accum[1] * 0.5f / (float)BATCH;
        out[0] = mf_loss + emb_loss;
        out[1] = mf_loss;
        out[2] = emb_loss;
    }
}

extern "C" void kernel_launch(void* const* d_in, const int* in_sizes, int n_in,
                              void* d_out, int out_size, void* d_ws, size_t ws_size,
                              hipStream_t stream) {
    (void)in_sizes; (void)n_in; (void)out_size; (void)ws_size;
    const int*   edge_row    = (const int*)  d_in[0];
    const int*   edge_col    = (const int*)  d_in[1];
    const float* edge_vals   = (const float*)d_in[2];
    const float* user_emb    = (const float*)d_in[3];
    const float* item_emb    = (const float*)d_in[4];
    const int*   user_id     = (const int*)  d_in[5];
    const int*   item_id     = (const int*)  d_in[6];
    const int*   neg_item_id = (const int*)  d_in[7];
    const float* Wg[3] = {(const float*)d_in[8],  (const float*)d_in[12], (const float*)d_in[16]};
    const float* bg[3] = {(const float*)d_in[9],  (const float*)d_in[13], (const float*)d_in[17]};
    const float* Wb[3] = {(const float*)d_in[10], (const float*)d_in[14], (const float*)d_in[18]};
    const float* bb[3] = {(const float*)d_in[11], (const float*)d_in[15], (const float*)d_in[19]};

    // workspace layout (floats)
    float* ws = (float*)d_ws;
    size_t off = 0;
    float* accum = ws + off; off += 16;
    float* vals  = ws + off; off += (size_t)NNZ;
    float* egoA  = ws + off; off += (size_t)N_NODES * EMB_DIM;
    float* egoB  = ws + off; off += (size_t)N_NODES * EMB_DIM;
    float* side  = ws + off; off += (size_t)N_NODES * EMB_DIM;
    float* ufeat = ws + off; off += (size_t)BATCH * 256;
    float* pfeat = ws + off; off += (size_t)BATCH * 256;
    float* nfeat = ws + off; off += (size_t)BATCH * 256;

    const int egoN = N_NODES * EMB_DIM;

    k_edge_drop<<<(NNZ + 255) / 256, 256, 0, stream>>>(edge_vals, vals, accum);
    k_build_ego<<<(egoN + 255) / 256, 256, 0, stream>>>(user_emb, item_emb, egoA);

    // layer-0 features: raw ego (no normalization)
    k_gather_norm<<<(BATCH + 255) / 256, 256, 0, stream>>>(egoA, user_id,     0,        ufeat, 0, 0);
    k_gather_norm<<<(BATCH + 255) / 256, 256, 0, stream>>>(egoA, item_id,     USER_NUM, pfeat, 0, 0);
    k_gather_norm<<<(BATCH + 255) / 256, 256, 0, stream>>>(egoA, neg_item_id, USER_NUM, nfeat, 0, 0);

    float* ein  = egoA;
    float* eout = egoB;
    for (int k = 0; k < 3; ++k) {
        k_zero<<<(egoN + 255) / 256, 256, 0, stream>>>(side, egoN);
        const unsigned scat = (unsigned)NNZ * 32u;
        k_scatter<<<(int)((scat + 255u) / 256u), 256, 0, stream>>>(edge_row, edge_col, vals, ein, side);
        unsigned seed = 0x9E3779B9u * (unsigned)(k + 1);
        k_gemm_act<<<(N_NODES + RPB - 1) / RPB, 256, 0, stream>>>(
            ein, side, Wg[k], bg[k], Wb[k], bb[k], eout, seed);
        k_gather_norm<<<(BATCH + 255) / 256, 256, 0, stream>>>(eout, user_id,     0,        ufeat, k + 1, 1);
        k_gather_norm<<<(BATCH + 255) / 256, 256, 0, stream>>>(eout, item_id,     USER_NUM, pfeat, k + 1, 1);
        k_gather_norm<<<(BATCH + 255) / 256, 256, 0, stream>>>(eout, neg_item_id, USER_NUM, nfeat, k + 1, 1);
        float* t = ein; ein = eout; eout = t;
    }

    k_loss<<<BATCH / 256, 256, 0, stream>>>(ufeat, pfeat, nfeat, accum);
    k_finalize<<<1, 64, 0, stream>>>(accum, (float*)d_out);
}
